// SLPMultFilteredNormalized_53919019434001
// MI455X (gfx1250) — matrix operand order repaired and checked by reading, not compiled, not run on hardware
//
#include <hip/hip_runtime.h>
#include <hip/hip_bf16.h>
#include <stdint.h>

// Problem constants (match reference)
#define B_DIM 4096
#define N_DIM 4096
#define M_DIM 16384

typedef __attribute__((ext_vector_type(16))) __bf16 v16bf;
typedef __attribute__((ext_vector_type(8)))  float  v8f;

union FragBF  { uint4 q[2]; v16bf v; };
union BF8Pack { __bf16 h[8]; uint4 q; };

// ---------------------------------------------------------------------------
// Async global->LDS copy (CDNA5 GLOBAL_LOAD_ASYNC_TO_LDS_B128, ASYNCcnt)
// with synchronous VGPR-staging fallback when the builtins are absent.
// Builtin signature (from hipcc diagnostics): pointers are to
// __attribute__((vector_size(16))) int, global src in AS1, LDS dst in AS3.
// ---------------------------------------------------------------------------
typedef int av4i __attribute__((vector_size(16)));

#if defined(__AMDGCN__) && \
    __has_builtin(__builtin_amdgcn_global_load_async_to_lds_b128) && \
    __has_builtin(__builtin_amdgcn_s_wait_asynccnt)
#define HAVE_ASYNC_LDS 1
#define ASYNC_CP(ldst, gsrc)                                                  \
    __builtin_amdgcn_global_load_async_to_lds_b128(                           \
        (__attribute__((address_space(1))) av4i*)(gsrc),                      \
        (__attribute__((address_space(3))) av4i*)(ldst), 0, 0)
#define ASYNC_WAIT() __builtin_amdgcn_s_wait_asynccnt(0)
#else
#define HAVE_ASYNC_LDS 0
#define ASYNC_CP(ldst, gsrc) (*(uint4*)(ldst) = *(const uint4*)(gsrc))
#define ASYNC_WAIT() ((void)0)
#endif

// ---------------------------------------------------------------------------
// f32 -> bf16 conversion, 8 elements / thread (two float4 loads, one b128 store)
// ---------------------------------------------------------------------------
__global__ __launch_bounds__(256)
void cvt_f32_to_bf16(const float* __restrict__ src, __bf16* __restrict__ dst,
                     long long n8) {
    long long i = (long long)blockIdx.x * blockDim.x + threadIdx.x;
    if (i >= n8) return;
    const float4* s = (const float4*)src + i * 2;
    float4 a = s[0];
    float4 b = s[1];
    BF8Pack p;
    p.h[0] = (__bf16)a.x; p.h[1] = (__bf16)a.y;
    p.h[2] = (__bf16)a.z; p.h[3] = (__bf16)a.w;
    p.h[4] = (__bf16)b.x; p.h[5] = (__bf16)b.y;
    p.h[6] = (__bf16)b.z; p.h[7] = (__bf16)b.w;
    ((uint4*)dst)[i] = p.q;
}

// ---------------------------------------------------------------------------
// Tiled WMMA GEMM:  C[r, c] = sum_k A[r, k] * Wt[c, k] + bias[c]
//   A  : row-major [rows, K]  bf16
//   Wt : row-major [cols, K]  bf16  (PyTorch W layout, so C = A @ W^T)
//
// Block = 256 threads = 8 waves; block tile 128(M) x 256(N), K-step 32.
// Wave grid 2(M) x 4(N): each wave owns 64x64 = 4x4 WMMA accumulators
// -> 16 ds_load_b128 feed 16 v_wmma per K-step (1:1).
// Double-buffered LDS; async global->LDS loads overlap WMMA compute.
// ---------------------------------------------------------------------------
template <bool FUSE_RELU_BF16>
__global__ __launch_bounds__(256)
void wmma_gemm_bt(const __bf16* __restrict__ A, const __bf16* __restrict__ Wt,
                  const float* __restrict__ bias, void* __restrict__ C,
                  int K, int ldc) {
    const int tid   = threadIdx.x;
    const int lane  = tid & 31;
    const int wid   = tid >> 5;
    const int waveM = wid & 1;   // 0..1 -> 64-row strip
    const int waveN = wid >> 1;  // 0..3 -> 64-col strip
    const int tileM = blockIdx.y * 128;
    const int tileN = blockIdx.x * 256;

    // Padded LDS tiles: row stride 40 bf16 = 80 bytes (16B aligned)
    __shared__ __bf16 As[2][128][40];
    __shared__ __bf16 Bs[2][256][40];

    v8f acc[4][4];
    const v8f vzero = {0.f, 0.f, 0.f, 0.f, 0.f, 0.f, 0.f, 0.f};
    for (int mi = 0; mi < 4; ++mi)
        for (int ni = 0; ni < 4; ++ni)
            acc[mi][ni] = vzero;

    // Cooperative tile-load indices:
    //  A tile 128x32 bf16 (8 KB):  row = tid/2, 16-elem half = tid%2
    //  B tile 256x32 bf16 (16 KB): row = tid, full 32-elem row (4 x b128)
    const int alr = tid >> 1;
    const int alh = (tid & 1) * 16;
    const int blr = tid;
    const __bf16* aG = A  + (size_t)(tileM + alr) * (size_t)K + alh;
    const __bf16* bG = Wt + (size_t)(tileN + blr) * (size_t)K;

    auto issue = [&](int buf, int k0) {
        ASYNC_CP(&As[buf][alr][alh],     aG + k0);
        ASYNC_CP(&As[buf][alr][alh + 8], aG + k0 + 8);
        ASYNC_CP(&Bs[buf][blr][0],       bG + k0);
        ASYNC_CP(&Bs[buf][blr][8],       bG + k0 + 8);
        ASYNC_CP(&Bs[buf][blr][16],      bG + k0 + 16);
        ASYNC_CP(&Bs[buf][blr][24],      bG + k0 + 24);
    };

    // WMMA fragment indexing (ISA 16-bit layouts, wave32)
    const int fr = lane & 15;  // M (A) / N (B) within 16x16 sub-tile
    const int hi = lane >> 4;  // lane-half selects K group

    issue(0, 0);
    int buf = 0;
    for (int k0 = 0; k0 < K; k0 += 32) {
        ASYNC_WAIT();          // this wave's async loads for `buf` landed
        __syncthreads();       // -> everyone's loads for `buf` landed
        if (k0 + 32 < K) issue(buf ^ 1, k0 + 32);

        // B fragments: lane holds col N=fr, K = {hi*16 .. +15} (contiguous)
        FragBF bfr[4];
        for (int ni = 0; ni < 4; ++ni) {
            const int c  = waveN * 64 + ni * 16 + fr;
            const int kb = hi * 16;
            bfr[ni].q[0] = *(const uint4*)&Bs[buf][c][kb];
            bfr[ni].q[1] = *(const uint4*)&Bs[buf][c][kb + 8];
        }
        // A fragments: lane holds row M=fr, K = {hi*16 .. +15} (contiguous)
        for (int mi = 0; mi < 4; ++mi) {
            FragBF afr;
            const int r  = waveM * 64 + mi * 16 + fr;
            const int kb = hi * 16;
            afr.q[0] = *(const uint4*)&As[buf][r][kb];
            afr.q[1] = *(const uint4*)&As[buf][r][kb + 8];
            for (int ni = 0; ni < 4; ++ni)
                acc[mi][ni] = __builtin_amdgcn_wmma_f32_16x16x32_bf16(
                    /*neg_a=*/false, afr.v,
                    /*neg_b=*/false, bfr[ni].v,
                    /*c_mod=*/(short)0, acc[mi][ni],
                    /*reuse_a=*/false, /*reuse_b=*/false);
        }
        buf ^= 1;
        // No trailing barrier needed: each wave's ds_loads are dscnt-waited
        // before its WMMAs (program order), hence before it signals the
        // next iteration's barrier; writes only ever target the other buffer.
    }

    // Epilogue: C 16x16 f32 layout -> VGPR v: (M = v + hi*8, N = fr)
    const int colBase = tileN + waveN * 64;
    const int rowBase = tileM + waveM * 64;
    for (int mi = 0; mi < 4; ++mi) {
        for (int ni = 0; ni < 4; ++ni) {
            const int gc = colBase + ni * 16 + fr;
            const float bv = bias[gc];
            const int r0 = rowBase + mi * 16 + hi * 8;
            for (int v = 0; v < 8; ++v) {
                float val = acc[mi][ni][v] + bv;
                const size_t idx = (size_t)(r0 + v) * (size_t)ldc + gc;
                if (FUSE_RELU_BF16) {
                    ((__bf16*)C)[idx] = (__bf16)(val > 0.f ? val : 0.f);
                } else {
                    ((float*)C)[idx] = val;
                }
            }
        }
    }
}

// ---------------------------------------------------------------------------
// Fused gate + masked row-normalize. One block (256 threads) per row, N=4096.
// out = mask ? (gA*(x*f) + gB*x) / s : 0 ; s = sum of masked gated (0 -> 1)
// ---------------------------------------------------------------------------
__global__ __launch_bounds__(256)
void gate_norm_rows(const float* __restrict__ x, const float* __restrict__ f,
                    const float* __restrict__ gA_p, const float* __restrict__ gB_p,
                    float* __restrict__ out) {
    const int row = blockIdx.x;
    const int tid = threadIdx.x;
    const float gA = gA_p[0];
    const float gB = gB_p[0];
    const float* xr = x + (size_t)row * N_DIM;
    const float* fr = f + (size_t)row * N_DIM;
    float*      orow = out + (size_t)row * N_DIM;

    float xs[16], gs[16];
    float partial = 0.f;
    for (int i = 0; i < 16; ++i) {
        const int j = tid + i * 256;
        const float xv = xr[j];
        const float fv = fr[j];
        const float g = gA * (xv * fv) + gB * xv;
        xs[i] = xv;
        gs[i] = g;
        if (xv != 0.f) partial += g;
    }

    __shared__ float red[256];
    red[tid] = partial;
    __syncthreads();
    for (int off = 128; off > 0; off >>= 1) {
        if (tid < off) red[tid] += red[tid + off];
        __syncthreads();
    }
    float s = red[0];
    if (s == 0.f) s = 1.f;
    const float inv = 1.f / s;

    for (int i = 0; i < 16; ++i) {
        const int j = tid + i * 256;
        orow[j] = (xs[i] != 0.f) ? gs[i] * inv : 0.f;
    }
}

// ---------------------------------------------------------------------------
// Launch: convert x/W1/W2 -> bf16, GEMM1 (relu, bf16 h), GEMM2 (f32 f),
// fused gate+normalize epilogue.
// Workspace layout (bytes):
//   xb  : B*N*2   =  32 MB   @ 0
//   w1b : M*N*2   = 128 MB   @ 32 MB
//   w2b : N*M*2   = 128 MB   @ 160 MB
//   h   : B*M*2   = 128 MB   @ 288 MB
//   f   : B*N*4   =  64 MB   @ 416 MB   (total 480 MB)
// ---------------------------------------------------------------------------
extern "C" void kernel_launch(void* const* d_in, const int* in_sizes, int n_in,
                              void* d_out, int out_size, void* d_ws, size_t ws_size,
                              hipStream_t stream) {
    const float* x  = (const float*)d_in[0];
    const float* W1 = (const float*)d_in[1];
    const float* b1 = (const float*)d_in[2];
    const float* W2 = (const float*)d_in[3];
    const float* b2 = (const float*)d_in[4];
    const float* gA = (const float*)d_in[5];
    const float* gB = (const float*)d_in[6];
    float* out = (float*)d_out;

    char* ws = (char*)d_ws;
    const size_t MB = 1024ull * 1024ull;
    __bf16* xb  = (__bf16*)(ws + 0);
    __bf16* w1b = (__bf16*)(ws + 32 * MB);
    __bf16* w2b = (__bf16*)(ws + 160 * MB);
    __bf16* h   = (__bf16*)(ws + 288 * MB);
    float*  f   = (float*) (ws + 416 * MB);

    // 1) conversions to bf16
    {
        long long n8 = (long long)B_DIM * N_DIM / 8;
        cvt_f32_to_bf16<<<(unsigned)((n8 + 255) / 256), 256, 0, stream>>>(x, xb, n8);
    }
    {
        long long n8 = (long long)M_DIM * N_DIM / 8;
        cvt_f32_to_bf16<<<(unsigned)((n8 + 255) / 256), 256, 0, stream>>>(W1, w1b, n8);
    }
    {
        long long n8 = (long long)N_DIM * M_DIM / 8;
        cvt_f32_to_bf16<<<(unsigned)((n8 + 255) / 256), 256, 0, stream>>>(W2, w2b, n8);
    }

    // 2) GEMM1: h = relu(x @ W1^T + b1)   [B, M] bf16
    {
        dim3 grid(M_DIM / 256, B_DIM / 128);
        wmma_gemm_bt<true><<<grid, 256, 0, stream>>>(xb, w1b, b1, (void*)h,
                                                     N_DIM, M_DIM);
    }

    // 3) GEMM2: f = h @ W2^T + b2         [B, N] f32
    {
        dim3 grid(N_DIM / 256, B_DIM / 128);
        wmma_gemm_bt<false><<<grid, 256, 0, stream>>>(h, w2b, b2, (void*)f,
                                                      M_DIM, N_DIM);
    }

    // 4) gate + masked row-normalize
    gate_norm_rows<<<B_DIM, 256, 0, stream>>>(x, f, gA, gB, out);
}
